// Discriminator_fgan2_43516608643120
// MI455X (gfx1250) — matrix operand order repaired and checked by reading, not compiled, not run on hardware
//
#include <hip/hip_runtime.h>
#include <hip/hip_bf16.h>
#include <math.h>

typedef __attribute__((ext_vector_type(16))) _Float16 v16h;
typedef __attribute__((ext_vector_type(8)))  _Float16 v8h;
typedef __attribute__((ext_vector_type(8)))  float    v8f;

#define NEG_SLOPE 0.2f
#define BN_EPS    1e-5f

// ---------------------------------------------------------------------------
// Stage 1: direct conv (Cin=1, 3x3, stride 2, VALID) + LeakyReLU
// out layout: [N][Cout][Ho][Ho]
// ---------------------------------------------------------------------------
__global__ void __launch_bounds__(256)
conv1_lrelu(const float* __restrict__ x, const float* __restrict__ w,
            const float* __restrict__ b, float* __restrict__ out,
            int N, int Hin, int Cout, int Ho) {
  int idx = blockIdx.x * blockDim.x + threadIdx.x;
  int total = N * Cout * Ho * Ho;
  if (idx >= total) return;
  int wo = idx % Ho; int t = idx / Ho;
  int ho = t % Ho;   t /= Ho;
  int c  = t % Cout; int n = t / Cout;
  const float* xp = x + (size_t)n * Hin * Hin + (size_t)(2 * ho) * Hin + 2 * wo;
  const float* wp = w + c * 9;
  float acc = b[c];
#pragma unroll
  for (int ky = 0; ky < 3; ++ky)
#pragma unroll
    for (int kx = 0; kx < 3; ++kx)
      acc += xp[ky * Hin + kx] * wp[ky * 3 + kx];
  out[idx] = acc >= 0.f ? acc : NEG_SLOPE * acc;
}

// ---------------------------------------------------------------------------
// 3x3 stride-2 VALID average pool (guide pyramid), 1 channel
// ---------------------------------------------------------------------------
__global__ void __launch_bounds__(256)
avgpool3s2(const float* __restrict__ in, float* __restrict__ out, int N, int Hin, int Ho) {
  int idx = blockIdx.x * blockDim.x + threadIdx.x;
  int total = N * Ho * Ho;
  if (idx >= total) return;
  int wo = idx % Ho; int t = idx / Ho;
  int ho = t % Ho;   int n = t / Ho;
  const float* ip = in + (size_t)n * Hin * Hin + (size_t)(2 * ho) * Hin + 2 * wo;
  float s = 0.f;
#pragma unroll
  for (int ky = 0; ky < 3; ++ky)
#pragma unroll
    for (int kx = 0; kx < 3; ++kx)
      s += ip[ky * Hin + kx];
  out[idx] = s * (1.0f / 9.0f);
}

// ---------------------------------------------------------------------------
// PAC gaussian kernel precompute: ker[n][kk][ho][wo] = exp(-0.5*(g - g_ctr)^2)
// ---------------------------------------------------------------------------
__global__ void __launch_bounds__(256)
pac_gauss(const float* __restrict__ guide, float* __restrict__ ker,
          int N, int Hg, int Ho, int Wo) {
  int idx = blockIdx.x * blockDim.x + threadIdx.x;
  int total = N * 9 * Ho * Wo;
  if (idx >= total) return;
  int wo = idx % Wo; int t = idx / Wo;
  int ho = t % Ho;   t /= Ho;
  int kk = t % 9;    int n = t / 9;
  int ky = kk / 3, kx = kk - 3 * ky;
  const float* gp = guide + (size_t)n * Hg * Hg;
  float g  = gp[(size_t)(2 * ho + ky) * Hg + (2 * wo + kx)];
  float gc = gp[(size_t)(2 * ho + 1)  * Hg + (2 * wo + 1)];
  float d = g - gc;
  ker[idx] = __expf(-0.5f * d * d);
}

// ---------------------------------------------------------------------------
// One-shot weight f32 -> f16 conversion (weights reused by every M-tile,
// so pay the conversion once and let the GEMM load WMMA-native f16).
// ---------------------------------------------------------------------------
__global__ void __launch_bounds__(256)
cvt_f32_to_f16(const float* __restrict__ in, _Float16* __restrict__ out, int n) {
  int idx = blockIdx.x * blockDim.x + threadIdx.x;
  if (idx < n) out[idx] = (_Float16)in[idx];
}

// ---------------------------------------------------------------------------
// PAC-conv as WMMA GEMM, fully shape-specialized.
//   A[m][k]  = xin[n][c][2ho+ky][2wo+kx] * ker[n][kk][ho][wo]  (k = c*9+kk)
//   B[k][nn] = wh[nn][k]   (pre-converted f16)
// One wave per 16(M) x 64(N) output block: one A fragment feeds 4 WMMAs.
// K loop fully unrolls (K/32 = 9/18/36); all index math constant-folds and
// the 9 per-lane ker loads CSE across the whole K sweep. B fragments are
// two aligned 16-byte f16 vector loads per tile (no conversions).
// A/B use contiguous K (k0 + 16*hiHalf + s) in both operands; C/D uses the ISA layout (wave32).
// ---------------------------------------------------------------------------
template <int CIN, int HIN, int COUT, int HO>
__global__ void __launch_bounds__(32)
pacconv_wmma(const float* __restrict__ xin, const float* __restrict__ ker,
             const _Float16* __restrict__ wh, const float* __restrict__ bias,
             float* __restrict__ out) {
  constexpr int K  = CIN * 9;
  constexpr int HW = HO * HO;

  const int lane   = threadIdx.x;   // 0..31
  const int mrow   = lane & 15;
  const int hiHalf = lane >> 4;     // 0 or 1

  // Pixel for this lane's A row (row M = blockIdx.x*16 + mrow)
  int m  = blockIdx.x * 16 + mrow;
  int n  = m / HW;                  // compile-time divisor
  int p  = m - n * HW;
  int ho = p / HO;
  int wo = p - ho * HO;
  const float* __restrict__ xp = xin + (size_t)n * CIN * HIN * HIN
                                     + (size_t)(2 * ho) * HIN + 2 * wo;
  const float* __restrict__ kp = ker + (size_t)n * 9 * HW + (size_t)ho * HO + wo;

  // 4 B columns (output channels) handled by this lane
  const int c0 = blockIdx.y * 64;
  const _Float16* __restrict__ wr0 = wh + (size_t)(c0 + 0  + mrow) * K;
  const _Float16* __restrict__ wr1 = wh + (size_t)(c0 + 16 + mrow) * K;
  const _Float16* __restrict__ wr2 = wh + (size_t)(c0 + 32 + mrow) * K;
  const _Float16* __restrict__ wr3 = wh + (size_t)(c0 + 48 + mrow) * K;
  __builtin_prefetch(wr0, 0, 3);
  __builtin_prefetch(wr1, 0, 3);
  __builtin_prefetch(wr2, 0, 3);
  __builtin_prefetch(wr3, 0, 3);

  v8f acc0 = {}, acc1 = {}, acc2 = {}, acc3 = {};

#pragma unroll
  for (int k0 = 0; k0 < K; k0 += 32) {
    // ---- A fragment: 16x32 f16, contiguous K per lane half ----
    // lane<16 : halfs 0-15 -> K=k0+0..15
    // lane>=16: halfs 0-15 -> K=k0+16..31
    v16h a;
#pragma unroll
    for (int g = 0; g < 2; ++g) {
      int kg = k0 + hiHalf * 16 + g * 8;
      int c  = kg / 9;                // constant-folds after unroll
      int kk = kg - 9 * c;
#pragma unroll
      for (int h = 0; h < 8; ++h) {
        int ky = kk / 3, kx = kk - 3 * ky;
        float xv = xp[c * HIN * HIN + ky * HIN + kx];
        float kv = kp[kk * HW];       // only 9 distinct loads over full K sweep
        a[g * 8 + h] = (_Float16)(xv * kv);
        if (++kk == 9) { kk = 0; ++c; }
      }
    }
    // ---- B fragments: 32x16 f16; lane holds 16 consecutive K values ----
    // 16-byte aligned vector loads (rows are 64B aligned, kb is a 32B multiple)
    const int kb = k0 + (hiHalf << 4);
    const v8h* p0 = (const v8h*)(wr0 + kb);
    const v8h* p1 = (const v8h*)(wr1 + kb);
    const v8h* p2 = (const v8h*)(wr2 + kb);
    const v8h* p3 = (const v8h*)(wr3 + kb);
    v16h b0 = __builtin_shufflevector(p0[0], p0[1], 0,1,2,3,4,5,6,7,8,9,10,11,12,13,14,15);
    v16h b1 = __builtin_shufflevector(p1[0], p1[1], 0,1,2,3,4,5,6,7,8,9,10,11,12,13,14,15);
    v16h b2 = __builtin_shufflevector(p2[0], p2[1], 0,1,2,3,4,5,6,7,8,9,10,11,12,13,14,15);
    v16h b3 = __builtin_shufflevector(p3[0], p3[1], 0,1,2,3,4,5,6,7,8,9,10,11,12,13,14,15);

    acc0 = __builtin_amdgcn_wmma_f32_16x16x32_f16(false, a, false, b0, (short)0, acc0, false, false);
    acc1 = __builtin_amdgcn_wmma_f32_16x16x32_f16(false, a, false, b1, (short)0, acc1, false, false);
    acc2 = __builtin_amdgcn_wmma_f32_16x16x32_f16(false, a, false, b2, (short)0, acc2, false, false);
    acc3 = __builtin_amdgcn_wmma_f32_16x16x32_f16(false, a, false, b3, (short)0, acc3, false, false);
  }

  // ---- epilogue: +bias, LeakyReLU, scatter (C/D layout: VGPR r -> M=r+8*hiHalf, N=lane&15)
  const float bv0 = bias[c0 + 0  + mrow];
  const float bv1 = bias[c0 + 16 + mrow];
  const float bv2 = bias[c0 + 32 + mrow];
  const float bv3 = bias[c0 + 48 + mrow];
#pragma unroll
  for (int r = 0; r < 8; ++r) {
    int mm = blockIdx.x * 16 + r + (hiHalf << 3);
    int nn = mm / HW;
    int pp = mm - nn * HW;
    size_t base = (size_t)nn * COUT * HW + pp;
    float v0 = acc0[r] + bv0; v0 = v0 >= 0.f ? v0 : NEG_SLOPE * v0;
    float v1 = acc1[r] + bv1; v1 = v1 >= 0.f ? v1 : NEG_SLOPE * v1;
    float v2 = acc2[r] + bv2; v2 = v2 >= 0.f ? v2 : NEG_SLOPE * v2;
    float v3 = acc3[r] + bv3; v3 = v3 >= 0.f ? v3 : NEG_SLOPE * v3;
    out[base + (size_t)(c0 + 0  + mrow) * HW] = v0;
    out[base + (size_t)(c0 + 16 + mrow) * HW] = v1;
    out[base + (size_t)(c0 + 32 + mrow) * HW] = v2;
    out[base + (size_t)(c0 + 48 + mrow) * HW] = v3;
  }
}

// ---------------------------------------------------------------------------
// Train-mode BatchNorm stats (deterministic: fixed-stride + LDS tree)
// ---------------------------------------------------------------------------
__global__ void __launch_bounds__(256)
bn_stats(const float* __restrict__ x, float* __restrict__ mean, float* __restrict__ rstd,
         int N, int C, int HW) {
  __shared__ float ssum[256];
  __shared__ float ssq[256];
  int ch = blockIdx.x;
  int total = N * HW;
  float s = 0.f, q = 0.f;
  for (int i = threadIdx.x; i < total; i += 256) {
    int n = i / HW; int p = i - n * HW;
    float v = x[((size_t)n * C + ch) * HW + p];
    s += v; q += v * v;
  }
  ssum[threadIdx.x] = s; ssq[threadIdx.x] = q;
  __syncthreads();
  for (int off = 128; off > 0; off >>= 1) {
    if (threadIdx.x < off) {
      ssum[threadIdx.x] += ssum[threadIdx.x + off];
      ssq[threadIdx.x]  += ssq[threadIdx.x + off];
    }
    __syncthreads();
  }
  if (threadIdx.x == 0) {
    float inv = 1.0f / (float)total;
    float mu  = ssum[0] * inv;
    float var = ssq[0] * inv - mu * mu;
    mean[ch] = mu;
    rstd[ch] = rsqrtf(var + BN_EPS);
  }
}

__global__ void __launch_bounds__(256)
bn_apply(float* __restrict__ x, const float* __restrict__ mean, const float* __restrict__ rstd,
         const float* __restrict__ g, const float* __restrict__ be, int C, int HW, int total) {
  int idx = blockIdx.x * blockDim.x + threadIdx.x;
  if (idx >= total) return;
  int ch = (idx / HW) % C;
  x[idx] = (x[idx] - mean[ch]) * rstd[ch] * g[ch] + be[ch];
}

// ---------------------------------------------------------------------------
// EqualLinear head: out[n] = dot(flat[n], wadv) / sqrt(D) + badv   (D=9216)
// ---------------------------------------------------------------------------
__global__ void __launch_bounds__(256)
final_linear(const float* __restrict__ flat, const float* __restrict__ wadv,
             const float* __restrict__ badv, float* __restrict__ out, int D) {
  __shared__ float s[256];
  int n = blockIdx.x;
  float acc = 0.f;
  for (int i = threadIdx.x; i < D; i += 256)
    acc += flat[(size_t)n * D + i] * wadv[i];
  s[threadIdx.x] = acc;
  __syncthreads();
  for (int off = 128; off > 0; off >>= 1) {
    if (threadIdx.x < off) s[threadIdx.x] += s[threadIdx.x + off];
    __syncthreads();
  }
  if (threadIdx.x == 0)
    out[n] = s[0] * (1.0f / 96.0f) + badv[0];   // 1/sqrt(9216) = 1/96
}

// ---------------------------------------------------------------------------
// Host-side orchestration
// ---------------------------------------------------------------------------
extern "C" void kernel_launch(void* const* d_in, const int* in_sizes, int n_in,
                              void* d_out, int out_size, void* d_ws, size_t ws_size,
                              hipStream_t stream) {
  (void)in_sizes; (void)n_in; (void)out_size; (void)ws_size;
  const float* x     = (const float*)d_in[0];
  const float* guide = (const float*)d_in[1];
  const float* w1 = (const float*)d_in[2];  const float* b1 = (const float*)d_in[3];
  const float* w2 = (const float*)d_in[4];  const float* b2 = (const float*)d_in[5];
  const float* w3 = (const float*)d_in[6];  const float* b3 = (const float*)d_in[7];
  const float* w4 = (const float*)d_in[8];  const float* b4 = (const float*)d_in[9];
  const float* g2 = (const float*)d_in[10]; const float* be2 = (const float*)d_in[11];
  const float* g3 = (const float*)d_in[12]; const float* be3 = (const float*)d_in[13];
  const float* g4 = (const float*)d_in[14]; const float* be4 = (const float*)d_in[15];
  const float* wadv = (const float*)d_in[16];
  const float* badv = (const float*)d_in[17];
  float* outv = (float*)d_out;

  const int N = 128;
  const int H0 = 120, H1 = 59, H2 = 29, H3 = 14, H4 = 6;
  const int C1 = 32, C2 = 64, C3 = 128, C4 = 256;

  // Workspace layout (floats; every section is a multiple of 16 floats, so
  // all sections inherit the allocation's 64B+ alignment).
  float* ws = (float*)d_ws;
  size_t off = 0;
  float* out1   = ws + off; off += (size_t)N * C1 * H1 * H1;
  float* guide1 = ws + off; off += (size_t)N * H1 * H1;
  float* out2   = ws + off; off += (size_t)N * C2 * H2 * H2;
  float* guide2 = ws + off; off += (size_t)N * H2 * H2;
  float* out3   = ws + off; off += (size_t)N * C3 * H3 * H3;
  float* guide3 = ws + off; off += (size_t)N * H3 * H3;
  float* out4   = ws + off; off += (size_t)N * C4 * H4 * H4;
  float* kerb   = ws + off; off += (size_t)N * 9 * H2 * H2;
  float* bnmean = ws + off; off += 256;
  float* bnrstd = ws + off; off += 256;
  _Float16* w2h = (_Float16*)(ws + off); off += (size_t)C2 * C1 * 9 / 2;   //  18,432 halfs
  _Float16* w3h = (_Float16*)(ws + off); off += (size_t)C3 * C2 * 9 / 2;   //  73,728 halfs
  _Float16* w4h = (_Float16*)(ws + off); off += (size_t)C4 * C3 * 9 / 2;   // 294,912 halfs

  const int TB = 256;
  auto cdiv = [](int a, int b) { return (a + b - 1) / b; };

  // ---- One-shot: weights -> f16 (WMMA-native operand format)
  cvt_f32_to_f16<<<cdiv(C2 * C1 * 9, TB), TB, 0, stream>>>(w2, w2h, C2 * C1 * 9);
  cvt_f32_to_f16<<<cdiv(C3 * C2 * 9, TB), TB, 0, stream>>>(w3, w3h, C3 * C2 * 9);
  cvt_f32_to_f16<<<cdiv(C4 * C3 * 9, TB), TB, 0, stream>>>(w4, w4h, C4 * C3 * 9);

  // ---- Stage 1: conv1 + lrelu; guide avgpool 120->59
  {
    int tot = N * C1 * H1 * H1;
    conv1_lrelu<<<cdiv(tot, TB), TB, 0, stream>>>(x, w1, b1, out1, N, H0, C1, H1);
    int tg = N * H1 * H1;
    avgpool3s2<<<cdiv(tg, TB), TB, 0, stream>>>(guide, guide1, N, H0, H1);
  }
  // ---- Layer 2: pacconv 32->64 (59 -> 29), lrelu, BN; guide 59->29
  {
    int tk = N * 9 * H2 * H2;
    pac_gauss<<<cdiv(tk, TB), TB, 0, stream>>>(guide1, kerb, N, H1, H2, H2);
    int M = N * H2 * H2;                           // 107,648
    dim3 grid(M / 16, C2 / 64);
    pacconv_wmma<32, 59, 64, 29><<<grid, 32, 0, stream>>>(out1, kerb, w2h, b2, out2);
    bn_stats<<<C2, TB, 0, stream>>>(out2, bnmean, bnrstd, N, C2, H2 * H2);
    int tot = N * C2 * H2 * H2;
    bn_apply<<<cdiv(tot, TB), TB, 0, stream>>>(out2, bnmean, bnrstd, g2, be2, C2, H2 * H2, tot);
    int tg = N * H2 * H2;
    avgpool3s2<<<cdiv(tg, TB), TB, 0, stream>>>(guide1, guide2, N, H1, H2);
  }
  // ---- Layer 3: pacconv 64->128 (29 -> 14), lrelu, BN; guide 29->14
  {
    int tk = N * 9 * H3 * H3;
    pac_gauss<<<cdiv(tk, TB), TB, 0, stream>>>(guide2, kerb, N, H2, H3, H3);
    int M = N * H3 * H3;                           // 25,088
    dim3 grid(M / 16, C3 / 64);
    pacconv_wmma<64, 29, 128, 14><<<grid, 32, 0, stream>>>(out2, kerb, w3h, b3, out3);
    bn_stats<<<C3, TB, 0, stream>>>(out3, bnmean, bnrstd, N, C3, H3 * H3);
    int tot = N * C3 * H3 * H3;
    bn_apply<<<cdiv(tot, TB), TB, 0, stream>>>(out3, bnmean, bnrstd, g3, be3, C3, H3 * H3, tot);
    int tg = N * H3 * H3;
    avgpool3s2<<<cdiv(tg, TB), TB, 0, stream>>>(guide2, guide3, N, H2, H3);
  }
  // ---- Layer 4: pacconv 128->256 (14 -> 6), lrelu, BN
  {
    int tk = N * 9 * H4 * H4;
    pac_gauss<<<cdiv(tk, TB), TB, 0, stream>>>(guide3, kerb, N, H3, H4, H4);
    int M = N * H4 * H4;                           // 4,608
    dim3 grid(M / 16, C4 / 64);
    pacconv_wmma<128, 14, 256, 6><<<grid, 32, 0, stream>>>(out3, kerb, w4h, b4, out4);
    bn_stats<<<C4, TB, 0, stream>>>(out4, bnmean, bnrstd, N, C4, H4 * H4);
    int tot = N * C4 * H4 * H4;
    bn_apply<<<cdiv(tot, TB), TB, 0, stream>>>(out4, bnmean, bnrstd, g4, be4, C4, H4 * H4, tot);
  }
  // ---- Head: flatten (out4 is already [N][C4*H4*H4] contiguous) + EqualLinear
  final_linear<<<N, TB, 0, stream>>>(out4, wadv, badv, outv, C4 * H4 * H4);
}
